// MPA_39934605918849
// MI455X (gfx1250) — matrix operand order repaired and checked by reading, not compiled, not run on hardware
//
#include <hip/hip_runtime.h>
#include <math.h>

typedef __attribute__((ext_vector_type(16))) _Float16 v16h;
typedef __attribute__((ext_vector_type(8)))  float    v8f;

namespace {
constexpr int kN    = 118;
constexpr int kT    = 2048;
constexpr int kF    = 128;   // n_feat
constexpr int kDP   = 128;   // LSTM hidden
constexpr int kG4   = 512;   // 4*DP gates
constexpr int kNPad = 128;
constexpr int kDN   = 256;
constexpr float kNeg = -1.0e9f;

// d_out layout (floats): y_hat(236) h_pmt(236) l_ort(1) l_pol(1) V1(118) V0(118)
constexpr int Y_OFF  = 0;
constexpr int HP_OFF = 236;
constexpr int ORT_OFF = 472;
constexpr int POL_OFF = 473;
constexpr int V1_OFF = 474;
constexpr int V0_OFF = 592;

// LSTM LDS: sH + sX (f16) + gates + c + bias (f32)  == 51,200 B (weights now in VGPRs)
constexpr size_t kLstmSmem =
    (size_t)2 * 16 * kDP * sizeof(_Float16)
  + (size_t)16 * kG4 * sizeof(float)
  + (size_t)16 * kDP * sizeof(float)
  + (size_t)kG4 * sizeof(float);
} // namespace

__device__ __forceinline__ float lk(float v) { return v >= 0.f ? v : 0.01f * v; }
__device__ __forceinline__ float sg(float z) { return 1.f / (1.f + __expf(-z)); }

// A-matrix (16x32 f16) per-lane K map: lanes 0-15 hold K 0-15; lanes 16-31 hold K 16-31 (contiguous)
__device__ __forceinline__ int amap(int lane, int e) {
  int base = (lane & 16) ? 16 : 0;
  return base + e;
}
// B-matrix (32x16 f16): lanes 0-15 hold K 0-15, lanes 16-31 hold K 16-31 (contiguous)
__device__ __forceinline__ int bmap(int lane, int e) {
  return e + ((lane & 16) ? 16 : 0);
}

__device__ __forceinline__ float blockReduceSum(float v, float* red) {
  int tid = threadIdx.x;
  red[tid] = v; __syncthreads();
  for (int s = blockDim.x >> 1; s > 0; s >>= 1) {
    if (tid < s) red[tid] += red[tid + s];
    __syncthreads();
  }
  float r = red[0]; __syncthreads();
  return r;
}
__device__ __forceinline__ float blockReduceMax(float v, float* red) {
  int tid = threadIdx.x;
  red[tid] = v; __syncthreads();
  for (int s = blockDim.x >> 1; s > 0; s >>= 1) {
    if (tid < s) red[tid] = fmaxf(red[tid], red[tid + s]);
    __syncthreads();
  }
  float r = red[0]; __syncthreads();
  return r;
}

// ---------------------------------------------------------------------------
// Persistent LSTM layer kernel. grid = 8 blocks (16 node-rows each),
// block = 512 threads = 16 waves; each wave owns 2 of the 32 gate-column
// tiles and holds its 16 weight B-fragments (128 VGPRs) in registers for all
// 2048 steps. Per step per wave: 8 A-fragment LDS loads feed 16
// v_wmma_f32_16x16x32_f16 as two interleaved independent accumulator chains.
// ---------------------------------------------------------------------------
__global__ __launch_bounds__(512) void lstm_kernel(
    const float* __restrict__ x32,            // (N,T,F) or null
    const _Float16* __restrict__ xseq16,      // [T][128][128] or null
    const float* __restrict__ Wih, const float* __restrict__ Whh,   // (512,128)
    const float* __restrict__ bih, const float* __restrict__ bhh,   // (512)
    _Float16* __restrict__ hseq_out,          // [T][128][128] or null
    float* __restrict__ p_out)                // [128][128] or null
{
  extern __shared__ char smem[];
  _Float16* sH = (_Float16*)smem;            // 16 x 128
  _Float16* sX = sH + 16 * kDP;              // 16 x 128
  float* sGate = (float*)(sX + 16 * kDP);    // 16 x 512
  float* sC    = sGate + 16 * kG4;           // 16 x 128
  float* sBias = sC + 16 * kDP;              // 512

  const int tid  = threadIdx.x;
  const int wave = tid >> 5;
  const int lane = tid & 31;
  const int nrow = lane & 15;
  const int hi   = (lane >> 4) & 1;
  const int rowbase = blockIdx.x * 16;
  const int ct0 = wave * 2;                  // this wave's two gate-column tiles

  // ---- load this wave's weight B-fragments into registers (f32 -> f16) ----
  v16h wih[2][4], whh[2][4];
#pragma unroll
  for (int c = 0; c < 2; ++c) {
#pragma unroll
    for (int kk = 0; kk < 4; ++kk) {
      const int wrow = (ct0 + c) * 16 + nrow;
      const int kbase = kk * 32 + ((lane & 16) ? 16 : 0);
      const float* sih = &Wih[wrow * kDP + kbase];
      const float* shh = &Whh[wrow * kDP + kbase];
#pragma unroll
      for (int e = 0; e < 16; ++e) {
        wih[c][kk][e] = (_Float16)sih[e];
        whh[c][kk][e] = (_Float16)shh[e];
      }
    }
  }

  for (int idx = tid; idx < kG4; idx += 512) sBias[idx] = bih[idx] + bhh[idx];
  for (int idx = tid; idx < 16 * kDP; idx += 512) {
    sH[idx] = (_Float16)0.f;
    sC[idx] = 0.f;
  }
  __syncthreads();

  for (int t = 0; t < kT; ++t) {
    // stage x_t tile (16 x 128) into LDS (f16), prefetch t+1
    for (int idx = tid; idx < 16 * kDP; idx += 512) {
      int r = idx >> 7, k = idx & 127;
      int node = rowbase + r;
      _Float16 v;
      if (x32) {
        float f = (node < kN) ? x32[((size_t)node * kT + t) * kF + k] : 0.f;
        v = (_Float16)f;
        if (node < kN && t + 1 < kT)
          __builtin_prefetch(&x32[((size_t)node * kT + (t + 1)) * kF + k], 0, 1);
      } else {
        v = xseq16[((size_t)t * kNPad + node) * kDP + k];
        if (t + 1 < kT)
          __builtin_prefetch(&xseq16[((size_t)(t + 1) * kNPad + node) * kDP + k], 0, 1);
      }
      sX[idx] = v;
    }
    __syncthreads();

    // gates: two independent WMMA accumulator chains, weights from registers
    const float bv0 = sBias[ct0 * 16 + nrow];
    const float bv1 = sBias[(ct0 + 1) * 16 + nrow];
    v8f acc0, acc1;
#pragma unroll
    for (int r = 0; r < 8; ++r) { acc0[r] = bv0; acc1[r] = bv1; }
#pragma unroll
    for (int kk = 0; kk < 4; ++kk) {
      const int kb = kk * 32;
      v16h ax, ah;
#pragma unroll
      for (int e = 0; e < 16; ++e) {
        ax[e] = sX[nrow * kDP + kb + amap(lane, e)];
        ah[e] = sH[nrow * kDP + kb + amap(lane, e)];
      }
      acc0 = __builtin_amdgcn_wmma_f32_16x16x32_f16(false, ax, false, wih[0][kk],
                                                    (short)0, acc0, false, false);
      acc1 = __builtin_amdgcn_wmma_f32_16x16x32_f16(false, ax, false, wih[1][kk],
                                                    (short)0, acc1, false, false);
      acc0 = __builtin_amdgcn_wmma_f32_16x16x32_f16(false, ah, false, whh[0][kk],
                                                    (short)0, acc0, false, false);
      acc1 = __builtin_amdgcn_wmma_f32_16x16x32_f16(false, ah, false, whh[1][kk],
                                                    (short)0, acc1, false, false);
    }
#pragma unroll
    for (int r = 0; r < 8; ++r) {
      sGate[(r + 8 * hi) * kG4 + ct0 * 16 + nrow]       = acc0[r];
      sGate[(r + 8 * hi) * kG4 + (ct0 + 1) * 16 + nrow] = acc1[r];
    }
    __syncthreads();

    // elementwise gate nonlinearities + state update
    for (int idx = tid; idx < 16 * kDP; idx += 512) {
      int r = idx >> 7, u = idx & 127;
      float gi = sGate[r * kG4 + u];
      float gf = sGate[r * kG4 + 128 + u];
      float gg = sGate[r * kG4 + 256 + u];
      float go = sGate[r * kG4 + 384 + u];
      float c = sg(gf) * sC[idx] + sg(gi) * tanhf(gg);
      float h = sg(go) * tanhf(c);
      sC[idx] = c;
      sH[idx] = (_Float16)h;
      if (hseq_out)
        hseq_out[((size_t)t * kNPad + rowbase + r) * kDP + u] = (_Float16)h;
      if (p_out && t == kT - 1)
        p_out[(rowbase + r) * kDP + u] = lk(h);
    }
    __syncthreads();
  }
}

// ---------------------------------------------------------------------------
// Generic small GEMM: C[M,N] (f32) = leaky?(A[M,K] (f32) * B[K,N] (f32) + bias)
// One wave per 16x16 output tile; on-the-fly f32->f16 conversion; K/M/N guards.
// ---------------------------------------------------------------------------
__global__ __launch_bounds__(32) void gemm16_kernel(
    float* __restrict__ C, int ldc,
    const float* __restrict__ A, int lda,
    const float* __restrict__ B, int ldb,
    const float* __restrict__ bias,
    int M, int N, int K, int doleaky)
{
  const int lane = threadIdx.x & 31;
  const int nr = lane & 15;
  const int hi = (lane >> 4) & 1;
  const int row = blockIdx.x * 16 + nr;
  const int col = blockIdx.y * 16 + nr;
  v8f acc;
#pragma unroll
  for (int r = 0; r < 8; ++r) acc[r] = 0.f;
  for (int kb = 0; kb < K; kb += 32) {
    v16h a, b;
#pragma unroll
    for (int e = 0; e < 16; ++e) {
      int ka = kb + amap(lane, e);
      int kbb = kb + bmap(lane, e);
      a[e] = (_Float16)((row < M && ka < K) ? A[(size_t)row * lda + ka] : 0.f);
      b[e] = (_Float16)((col < N && kbb < K) ? B[(size_t)kbb * ldb + col] : 0.f);
    }
    acc = __builtin_amdgcn_wmma_f32_16x16x32_f16(false, a, false, b,
                                                 (short)0, acc, false, false);
  }
#pragma unroll
  for (int r = 0; r < 8; ++r) {
    int m = blockIdx.x * 16 + r + 8 * hi;
    int n = blockIdx.y * 16 + nr;
    if (m < M && n < N) {
      float v = acc[r];
      if (bias) v += bias[n];
      if (doleaky) v = lk(v);
      C[(size_t)m * ldc + n] = v;
    }
  }
}

// --------------------------- head kernels ----------------------------------
__global__ void prep_kernel(const float* __restrict__ x_tag,
                            const float* __restrict__ pm,
                            const float* __restrict__ Wvm,
                            float* __restrict__ act, float* __restrict__ vm2,
                            float* __restrict__ sumV1, float* __restrict__ out)
{
  int tid = threadIdx.x;
  __shared__ float sv[256];
  float v1 = 0.f;
  if (tid < kN) {
    float a = x_tag[tid * 3 + 2];
    act[tid] = a;
    out[HP_OFF + tid * 2 + 0] = x_tag[tid * 3 + 0];
    out[HP_OFF + tid * 2 + 1] = x_tag[tid * 3 + 1];
    float isV1 = (a == 0.f) ? 1.f : 0.f;
    float isV0 = (a == 1.f) ? 1.f : 0.f;
    out[V1_OFF + tid] = isV1;
    out[V0_OFF + tid] = isV0;
    v1 = isV1;
  }
  sv[tid] = v1;
  __syncthreads();
  for (int s = 128; s > 0; s >>= 1) { if (tid < s) sv[tid] += sv[tid + s]; __syncthreads(); }
  if (tid == 0) *sumV1 = sv[0];
  // v_m rows for act=0 and act=1: leaky([pm, act] @ Wvm), Wvm (769,128)
  {
    int a = tid >> 7;
    int r = tid & 127;
    float s = 0.f;
    for (int k = 0; k < 768; ++k) s += pm[k] * Wvm[k * 128 + r];
    s += (float)a * Wvm[768 * 128 + r];
    vm2[a * 128 + r] = lk(s);
  }
}

__global__ void hcat_kernel(const float* up, const float* vm2, const float* act,
                            float* hcat) {
  int idx = blockIdx.x * 256 + threadIdx.x;
  if (idx >= kN * 384) return;
  int i = idx / 384, c = idx % 384;
  int a = (act[i] != 0.f) ? 1 : 0;
  float v;
  if (c < 128)      v = up[i * 128 + c];
  else if (c < 256) v = up[i * 128 + (c - 128)] + vm2[a * 128 + (c - 128)];
  else              v = vm2[a * 128 + (c - 256)];
  hcat[idx] = v;
}

__global__ void hacat_kernel(const float* hyb, const float* x_tag, float* hacat) {
  int idx = blockIdx.x * 256 + threadIdx.x;
  if (idx >= kN * 258) return;
  int i = idx / 258, c = idx % 258;
  hacat[idx] = (c < 256) ? hyb[i * 256 + c] : x_tag[i * 3 + (c - 256)];
}

__global__ void nmat_kernel(const float* nact, const float* ninact,
                            const float* act, float* nmat) {
  int idx = blockIdx.x * 256 + threadIdx.x;
  if (idx >= kN * kDN) return;
  int i = idx / kDN;
  float a = act[i];
  nmat[idx] = lk(nact[idx] * (1.f - a) + ninact[idx] * a);
}

// phi[i,j] = leaky(U[j] + V[i]) . aphi   (block = i, thread = j)
__global__ void phi_kernel(const float* __restrict__ U, const float* __restrict__ V,
                           const float* __restrict__ aphi, float* __restrict__ phi) {
  __shared__ float sV[192];
  int i = blockIdx.x;
  for (int r = threadIdx.x; r < 192; r += 128) sV[r] = V[i * 192 + r];
  __syncthreads();
  int j = threadIdx.x;
  if (j < kN) {
    float s = 0.f;
    for (int r = 0; r < 192; ++r) {
      float z = U[j * 192 + r] + sV[r];
      s += lk(z) * aphi[r];
    }
    phi[i * kN + j] = s;
  }
}

// masked softmaxes over axis 0 (i) for each column j; block = j, thread = i
__global__ void alpha_kernel(const float* __restrict__ phi, const float* __restrict__ act,
                             const float* __restrict__ sumV1, float* __restrict__ alpha) {
  __shared__ float red[128];
  int j = blockIdx.x, i = threadIdx.x;
  bool valid = i < kN;
  float a = valid ? act[i] : -1.f;
  float v = valid ? phi[i * kN + j] : 0.f;
  float m1 = (valid && a == 0.f) ? v : kNeg;
  float m0 = (valid && a == 1.f) ? v : kNeg;
  float va = valid ? v : kNeg;
  float mx1 = blockReduceMax(m1, red);
  float e1 = valid ? __expf(m1 - mx1) : 0.f;
  float s1 = blockReduceSum(e1, red);
  float mx0 = blockReduceMax(m0, red);
  float e0 = valid ? __expf(m0 - mx0) : 0.f;
  float s0 = blockReduceSum(e0, red);
  float mxa = blockReduceMax(va, red);
  float ea = valid ? __expf(va - mxa) : 0.f;
  float sa = blockReduceSum(ea, red);
  if (valid) {
    float p1 = e1 / s1, p0 = e0 / s0, pa = ea / sa;
    float pmask = (a == 0.f) ? p1 : ((a == 1.f) ? p0 : v);
    alpha[i * kN + j] = (*sumV1 > 0.f) ? pmask : pa;
  }
}

// T1[i] = sum_j a[i,j]*leaky(A1[j]+A2[i]); t2[i] = sum_j a[i,j]*B1[j]; s_a[i]=sum_j a
__global__ void tsum_kernel(const float* __restrict__ alpha, const float* __restrict__ A1,
                            const float* __restrict__ A2, const float* __restrict__ B1,
                            float* __restrict__ T1, float* __restrict__ t2,
                            float* __restrict__ s_a) {
  int i = blockIdx.x, d = threadIdx.x;
  float a2d = A2[i * 256 + d];
  float acc1 = 0.f, acc2 = 0.f, sa = 0.f;
  for (int j = 0; j < kN; ++j) {
    float al = alpha[i * kN + j];
    float z = A1[j * 256 + d] + a2d;
    acc1 += al * lk(z);
    acc2 += al * B1[j * 256 + d];
    sa += al;
  }
  T1[i * 256 + d] = acc1;
  t2[i * 256 + d] = acc2;
  if (d == 0) s_a[i] = sa;
}

// per-node output head: y_hat softmax + polarity unit-vector contribution
__global__ void node_out_kernel(const float* __restrict__ nmat, const float* __restrict__ T1W,
                                const float* __restrict__ t2, const float* __restrict__ s_a,
                                const float* __restrict__ B2, const float* __restrict__ act,
                                const float* __restrict__ x_tag, const float* __restrict__ Wi,
                                const float* __restrict__ bi, float* __restrict__ out,
                                float* __restrict__ poltmp) {
  __shared__ float sout[768];
  __shared__ float red[256];
  int i = blockIdx.x, d = threadIdx.x;
  float a = act[i];
  float nv = nmat[i * 256 + d];
  float Mi = T1W[i * 256 + d] + t2[i * 256 + d] + s_a[i] * B2[i * 256 + d];
  float hm1 = lk((1.f - a) * Mi);
  float hm0 = lk(a * Mi);
  sout[d] = nv; sout[256 + d] = hm0; sout[512 + d] = hm1;
  __syncthreads();
  float y0p = 0.f, y1p = 0.f;
  for (int dd = d; dd < 768; dd += 256) {
    float ov = sout[dd];
    y0p += ov * Wi[((size_t)i * 768 + dd) * 2 + 0];
    y1p += ov * Wi[((size_t)i * 768 + dd) * 2 + 1];
  }
  float y0 = blockReduceSum(y0p, red);
  float y1 = blockReduceSum(y1p, red);
  float ns = blockReduceSum(nv * nv, red);
  float nrm = fmaxf(sqrtf(ns), 1e-8f);
  float hp0 = x_tag[i * 3 + 0], hp1 = x_tag[i * 3 + 1];
  float sgn = (hp1 > hp0) ? 1.f : ((hp1 < hp0) ? -1.f : 0.f);
  float sfac = (a == 0.f) ? sgn : 0.f;
  poltmp[i * 256 + d] = sfac * nv / nrm;
  if (d == 0) {
    y0 += bi[i * 2 + 0]; y1 += bi[i * 2 + 1];
    float m = fmaxf(y0, y1);
    float e0 = __expf(y0 - m), e1 = __expf(y1 - m);
    out[Y_OFF + i * 2 + 0] = e0 / (e0 + e1);
    out[Y_OFF + i * 2 + 1] = e1 / (e0 + e1);
  }
}

// partial[k] = sum_l (A^T A)_{kl} (B^T B)_{kl}  -> ||A B^T||_F^2 via trace trick
__global__ void gram_trace_kernel(const float* __restrict__ A, const float* __restrict__ B,
                                  int R, float* __restrict__ partial) {
  __shared__ float red[128];
  int k = blockIdx.x, l = threadIdx.x;
  float p = 0.f, q = 0.f;
  for (int r = 0; r < R; ++r) {
    p += A[r * 128 + k] * A[r * 128 + l];
    q += B[r * 128 + k] * B[r * 128 + l];
  }
  float v = blockReduceSum(p * q, red);
  if (l == 0) partial[k] = v;
}

__global__ void finalize_kernel(const float* __restrict__ poltmp,
                                const float* __restrict__ part1,
                                const float* __restrict__ part2,
                                float* __restrict__ out) {
  __shared__ float red[256];
  int d = threadIdx.x;
  float c = 0.f;
  for (int i = 0; i < kN; ++i) c += poltmp[i * 256 + d];
  float tot = blockReduceSum(c * c, red);
  if (d == 0) {
    out[POL_OFF] = tot;
    float s1 = 0.f, s2 = 0.f;
    for (int k = 0; k < 128; ++k) { s1 += part1[k]; s2 += part2[k]; }
    out[ORT_OFF] = sqrtf(s1) + sqrtf(s2);
  }
}

// ---------------------------------------------------------------------------
extern "C" void kernel_launch(void* const* d_in, const int* in_sizes, int n_in,
                              void* d_out, int out_size, void* d_ws, size_t ws_size,
                              hipStream_t stream) {
  (void)in_sizes; (void)n_in; (void)out_size; (void)ws_size;
  const float* x     = (const float*)d_in[0];
  const float* x_tag = (const float*)d_in[1];
  const float* W_ih0 = (const float*)d_in[2];
  const float* W_hh0 = (const float*)d_in[3];
  const float* b_ih0 = (const float*)d_in[4];
  const float* b_hh0 = (const float*)d_in[5];
  const float* W_ih1 = (const float*)d_in[6];
  const float* W_hh1 = (const float*)d_in[7];
  const float* b_ih1 = (const float*)d_in[8];
  const float* b_hh1 = (const float*)d_in[9];
  const float* pm    = (const float*)d_in[10];
  const float* Wum   = (const float*)d_in[11];
  const float* Wvm   = (const float*)d_in[12];
  const float* Wup   = (const float*)d_in[13];
  const float* Wvp   = (const float*)d_in[14];
  const float* W_hyb = (const float*)d_in[15];
  const float* b_hyb = (const float*)d_in[16];
  const float* W_act = (const float*)d_in[17];
  const float* b_act = (const float*)d_in[18];
  const float* W_in  = (const float*)d_in[19];
  const float* b_in  = (const float*)d_in[20];
  const float* Wphin = (const float*)d_in[21];
  const float* aphi  = (const float*)d_in[22];
  const float* Won   = (const float*)d_in[23];
  const float* Weo   = (const float*)d_in[24];
  const float* Wi    = (const float*)d_in[25];
  const float* bi    = (const float*)d_in[26];
  float* out = (float*)d_out;

  char* w = (char*)d_ws;
  auto take = [&](size_t bytes) -> char* {
    char* p = w;
    w += (bytes + 255) & ~(size_t)255;
    return p;
  };
  _Float16* hseq  = (_Float16*)take((size_t)kT * kNPad * kDP * sizeof(_Float16)); // 64 MB
  float* pbuf   = (float*)take((size_t)kNPad * kDP * sizeof(float));
  float* act    = (float*)take(kN * sizeof(float));
  float* vm2    = (float*)take(2 * 128 * sizeof(float));
  float* sumV1  = (float*)take(sizeof(float));
  float* up     = (float*)take((size_t)kN * 128 * sizeof(float));
  float* hcat   = (float*)take((size_t)kN * 384 * sizeof(float));
  float* hyb    = (float*)take((size_t)kN * 256 * sizeof(float));
  float* hacat  = (float*)take((size_t)kN * 258 * sizeof(float));
  float* nact   = (float*)take((size_t)kN * 256 * sizeof(float));
  float* ninact = (float*)take((size_t)kN * 256 * sizeof(float));
  float* nmat   = (float*)take((size_t)kN * 256 * sizeof(float));
  float* Ubuf   = (float*)take((size_t)kN * 192 * sizeof(float));
  float* Vbuf   = (float*)take((size_t)kN * 192 * sizeof(float));
  float* A1     = (float*)take((size_t)kN * 256 * sizeof(float));
  float* A2     = (float*)take((size_t)kN * 256 * sizeof(float));
  float* B1     = (float*)take((size_t)kN * 256 * sizeof(float));
  float* B2     = (float*)take((size_t)kN * 256 * sizeof(float));
  float* phib   = (float*)take((size_t)kN * kN * sizeof(float));
  float* alphab = (float*)take((size_t)kN * kN * sizeof(float));
  float* T1     = (float*)take((size_t)kN * 256 * sizeof(float));
  float* t2b    = (float*)take((size_t)kN * 256 * sizeof(float));
  float* s_a    = (float*)take(kN * sizeof(float));
  float* T1W    = (float*)take((size_t)kN * 256 * sizeof(float));
  float* poltmp = (float*)take((size_t)kN * 256 * sizeof(float));
  float* part1  = (float*)take(128 * sizeof(float));
  float* part2  = (float*)take(128 * sizeof(float));

  prep_kernel<<<1, 256, 0, stream>>>(x_tag, pm, Wvm, act, vm2, sumV1, out);

  // LSTM layer 0: x -> h1 sequence (f16), layer 1: h1 -> p = leaky(h2[:, -1])
  lstm_kernel<<<8, 512, kLstmSmem, stream>>>(x, nullptr, W_ih0, W_hh0, b_ih0, b_hh0,
                                             hseq, nullptr);
  lstm_kernel<<<8, 512, kLstmSmem, stream>>>(nullptr, hseq, W_ih1, W_hh1, b_ih1, b_hh1,
                                             nullptr, pbuf);

  // u_p = leaky(p @ Wup)
  gemm16_kernel<<<dim3(8, 8), 32, 0, stream>>>(up, 128, pbuf, 128, Wup, 128,
                                               nullptr, kN, 128, 128, 1);
  hcat_kernel<<<(kN * 384 + 255) / 256, 256, 0, stream>>>(up, vm2, act, hcat);
  // h_hyb = leaky([u_p, u_p+v_m, v_m] @ W_hyb + b_hyb)
  gemm16_kernel<<<dim3(8, 16), 32, 0, stream>>>(hyb, 256, hcat, 384, W_hyb, 256,
                                                b_hyb, kN, 256, 384, 1);
  hacat_kernel<<<(kN * 258 + 255) / 256, 256, 0, stream>>>(hyb, x_tag, hacat);
  gemm16_kernel<<<dim3(8, 16), 32, 0, stream>>>(nact, 256, hacat, 258, W_act, 256,
                                                b_act, kN, 256, 258, 0);
  gemm16_kernel<<<dim3(8, 16), 32, 0, stream>>>(ninact, 256, hyb, 256, W_in, 256,
                                                b_in, kN, 256, 256, 0);
  nmat_kernel<<<(kN * 256 + 255) / 256, 256, 0, stream>>>(nact, ninact, act, nmat);

  // pair@W decompositions: U/V (Wphin halves), A1/A2 (Won halves), B1/B2 (Weo rows)
  gemm16_kernel<<<dim3(8, 12), 32, 0, stream>>>(Ubuf, 192, nmat, 256, Wphin, 192,
                                                nullptr, kN, 192, 256, 0);
  gemm16_kernel<<<dim3(8, 12), 32, 0, stream>>>(Vbuf, 192, nmat, 256,
                                                Wphin + (size_t)256 * 192, 192,
                                                nullptr, kN, 192, 256, 0);
  gemm16_kernel<<<dim3(8, 16), 32, 0, stream>>>(A1, 256, nmat, 256, Won, 256,
                                                nullptr, kN, 256, 256, 0);
  gemm16_kernel<<<dim3(8, 16), 32, 0, stream>>>(A2, 256, nmat, 256,
                                                Won + (size_t)256 * 256, 256,
                                                nullptr, kN, 256, 256, 0);
  gemm16_kernel<<<dim3(8, 16), 32, 0, stream>>>(B1, 256, nmat, 256,
                                                Weo + (size_t)256 * 256, 256,
                                                nullptr, kN, 256, 256, 0);
  gemm16_kernel<<<dim3(8, 16), 32, 0, stream>>>(B2, 256, nmat, 256,
                                                Weo + (size_t)512 * 256, 256,
                                                nullptr, kN, 256, 256, 0);

  phi_kernel<<<kN, 128, 0, stream>>>(Ubuf, Vbuf, aphi, phib);
  alpha_kernel<<<kN, 128, 0, stream>>>(phib, act, sumV1, alphab);
  tsum_kernel<<<kN, 256, 0, stream>>>(alphab, A1, A2, B1, T1, t2b, s_a);
  gemm16_kernel<<<dim3(8, 16), 32, 0, stream>>>(T1W, 256, T1, 256, Weo, 256,
                                                nullptr, kN, 256, 256, 0);
  node_out_kernel<<<kN, 256, 0, stream>>>(nmat, T1W, t2b, s_a, B2, act, x_tag,
                                          Wi, bi, out, poltmp);
  gram_trace_kernel<<<128, 128, 0, stream>>>(Wum, Wvm, 769, part1);
  gram_trace_kernel<<<128, 128, 0, stream>>>(Wup, Wvp, 128, part2);
  finalize_kernel<<<1, 256, 0, stream>>>(poltmp, part1, part2, out);
}
